// Attention_10428180594840
// MI455X (gfx1250) — matrix operand order repaired and checked by reading, not compiled, not run on hardware
//
#include <hip/hip_runtime.h>
#include <hip/hip_bf16.h>
#include <math.h>

typedef __attribute__((ext_vector_type(16))) _Float16 v16h;
typedef __attribute__((ext_vector_type(8)))  _Float16 h8v;
typedef __attribute__((ext_vector_type(8)))  float    v8f;

#define DIM    1024
#define HEADS  16
#define DHEAD  64
#define SEQ    2048
#define BATCH  2
#define ROWS   (BATCH * SEQ)   /* 4096 */
#define KVLD   (2 * DHEAD)     /* 128  */
#define LN_EPS 1e-5f

// ---------------------------------------------------------------------------
// LayerNorm over rows of DIM floats; templated output type (f16 or f32).
// ---------------------------------------------------------------------------
template <typename OT>
__global__ void ln_rows(const float* __restrict__ x, const float* __restrict__ g,
                        OT* __restrict__ out) {
    const int row = blockIdx.x;
    const float* xr = x + (size_t)row * DIM;
    const int t = threadIdx.x;                 // 0..255
    float4 v = ((const float4*)xr)[t];
    float s  = v.x + v.y + v.z + v.w;
    float ss = v.x * v.x + v.y * v.y + v.z * v.z + v.w * v.w;
#pragma unroll
    for (int off = 16; off > 0; off >>= 1) {
        s  += __shfl_xor(s, off);
        ss += __shfl_xor(ss, off);
    }
    __shared__ float sm[8], sq[8];
    __shared__ float mean_s, rstd_s;
    const int w = t >> 5;
    if ((t & 31) == 0) { sm[w] = s; sq[w] = ss; }
    __syncthreads();
    if (t == 0) {
        float S = 0.f, SS = 0.f;
        for (int i = 0; i < 8; ++i) { S += sm[i]; SS += sq[i]; }
        float mean = S * (1.0f / DIM);
        float var  = SS * (1.0f / DIM) - mean * mean;
        mean_s = mean;
        rstd_s = rsqrtf(var + LN_EPS);
    }
    __syncthreads();
    const float mean = mean_s, rstd = rstd_s;
    float4 gv = ((const float4*)g)[t];
    OT* orow = out + (size_t)row * DIM;
    orow[t * 4 + 0] = (OT)((v.x - mean) * rstd * gv.x);
    orow[t * 4 + 1] = (OT)((v.y - mean) * rstd * gv.y);
    orow[t * 4 + 2] = (OT)((v.z - mean) * rstd * gv.z);
    orow[t * 4 + 3] = (OT)((v.w - mean) * rstd * gv.w);
}

__global__ void f32_to_f16(const float* __restrict__ in, _Float16* __restrict__ out, int n) {
    int i = blockIdx.x * blockDim.x + threadIdx.x;
    if (i < n) out[i] = (_Float16)in[i];
}

// ---------------------------------------------------------------------------
// C(MxN) = A(MxK) @ W(NxK)^T, f16 in, fp32 WMMA accumulation.
// Block 256 = 8 waves; block tile 128x64; wave tile 16x64.
// Weight tile (64 n x 32 k) staged per block per k-step via async
// global->LDS DMA, double buffered; B fragments read from LDS by all waves.
// All four B fragments are hoisted before the WMMA group so the four
// v_wmma issue back-to-back behind a single s_wait_dscnt.
// ---------------------------------------------------------------------------
template <typename OT>
__global__ void gemm_nt_wmma(const _Float16* __restrict__ A,
                             const _Float16* __restrict__ W,
                             OT* __restrict__ C, int M, int N, int K) {
    __shared__ __align__(32) _Float16 wl[2][64 * 32];   // [n][k], double buffered

    const int tid  = threadIdx.x;
    const int lane = tid & 31;
    const int w    = tid >> 5;
    const int m0   = blockIdx.y * 128 + w * 16;
    const int n0   = blockIdx.x * 64;
    const int mr   = lane & 15;
    const int hi   = lane >> 4;
    const int kba  = hi ? 16 : 0;   // A-fragment K base (contiguous k order)
    const int kbb  = hi ? 16 : 0;   // B-fragment K base

    const int sn  = tid >> 2;          // 0..63  (4 threads per weight row)
    const int sk0 = (tid & 3) * 8;     // 0,8,16,24

    v8f acc[4] = {};
    const _Float16* arow = A + (size_t)(m0 + mr) * K;
    const _Float16* wbase = W + (size_t)(n0 + sn) * K + sk0;

    {   // prologue: stage k0 = 0
        uint64_t g = (uint64_t)(uintptr_t)(wbase);
        uint32_t l = (uint32_t)(uintptr_t)(&wl[0][sn * 32 + sk0]);
        asm volatile("global_load_async_to_lds_b128 %0, %1, off" :: "v"(l), "v"(g) : "memory");
    }

    int buf = 0;
    for (int k0 = 0; k0 < K; k0 += 32) {
        asm volatile("s_wait_asynccnt 0x0" ::: "memory");
        __syncthreads();
        if (k0 + 32 < K) {              // uniform branch: stage next tile
            uint64_t g = (uint64_t)(uintptr_t)(wbase + k0 + 32);
            uint32_t l = (uint32_t)(uintptr_t)(&wl[buf ^ 1][sn * 32 + sk0]);
            asm volatile("global_load_async_to_lds_b128 %0, %1, off" :: "v"(l), "v"(g) : "memory");
        }

        h8v alo = *(const h8v*)(arow + k0 + kba);
        h8v ahi = *(const h8v*)(arow + k0 + kba + 8);
        v16h a;
#pragma unroll
        for (int i = 0; i < 8; ++i) { a[i] = alo[i]; a[i + 8] = ahi[i]; }
        __builtin_prefetch(arow + k0 + 64, 0, 1);

        v16h bf[4];
#pragma unroll
        for (int t = 0; t < 4; ++t)
            bf[t] = *(const v16h*)&wl[buf][(t * 16 + mr) * 32 + kbb];
#pragma unroll
        for (int t = 0; t < 4; ++t)
            acc[t] = __builtin_amdgcn_wmma_f32_16x16x32_f16(
                false, a, false, bf[t], (short)0, acc[t], false, false);
        buf ^= 1;
    }
#pragma unroll
    for (int t = 0; t < 4; ++t)
#pragma unroll
        for (int j = 0; j < 8; ++j) {
            int row = m0 + j + hi * 8;
            int col = n0 + t * 16 + mr;
            C[(size_t)row * N + col] = (OT)acc[t][j];
        }
}

// k <- 4 * k/||k||; k = kv[:, 0:64], row stride 128. Block 256 = 8 rows.
__global__ void l2norm_k(_Float16* __restrict__ kv) {
    const int row  = blockIdx.x * 8 + (threadIdx.x >> 5);
    const int lane = threadIdx.x & 31;
    _Float16* p = kv + (size_t)row * KVLD + lane * 2;
    float a = (float)p[0], b = (float)p[1];
    float ss = a * a + b * b;
#pragma unroll
    for (int off = 16; off > 0; off >>= 1) ss += __shfl_xor(ss, off);
    float scale = 4.0f / fmaxf(sqrtf(ss), 1e-12f);
    p[0] = (_Float16)(a * scale);
    p[1] = (_Float16)(b * scale);
}

// ---------------------------------------------------------------------------
// Stage one 32-key tile: K rows via async global->LDS DMA (ASYNCcnt path),
// V rows via coalesced b128 load + transposed LDS store.
// ---------------------------------------------------------------------------
__device__ __forceinline__ void stage_kv(const _Float16* __restrict__ kbase,
                                         const _Float16* __restrict__ vbase,
                                         int j0, _Float16* kt_lds, _Float16* vt_lds,
                                         int tid) {
    const int tkey = tid >> 3;          // 0..31
    const int td0  = (tid & 7) * 8;     // 0,8,...,56
    uint64_t gk = (uint64_t)(uintptr_t)(kbase + (size_t)(j0 + tkey) * KVLD + td0);
    uint32_t lk = (uint32_t)(uintptr_t)(kt_lds + tkey * DHEAD + td0);
    asm volatile("global_load_async_to_lds_b128 %0, %1, off" :: "v"(lk), "v"(gk) : "memory");
    h8v vrow = *(const h8v*)(vbase + (size_t)(j0 + tkey) * KVLD + td0);
#pragma unroll
    for (int i = 0; i < 8; ++i) vt_lds[(td0 + i) * 32 + tkey] = vrow[i];  // V^T [d][key]
}

// ---------------------------------------------------------------------------
// Causal flash attention, transposed-S formulation, q-l2norm fused.
// Grid (SEQ/128, HEADS, BATCH); block 256 = 8 waves; wave = 16 queries.
// Fragment loads are hoisted ahead of each WMMA group so the matrix ops
// issue back-to-back behind a single LDS wait.
// ---------------------------------------------------------------------------
__global__ void flash_attn(const _Float16* __restrict__ q,
                           const _Float16* __restrict__ kv,
                           _Float16* __restrict__ o) {
    __shared__ __align__(32) _Float16 pls[8][16 * 32];      // per-wave P^T [q][key]
    __shared__ __align__(32) _Float16 ktile[2][32 * DHEAD]; // [key][d], double buffered
    __shared__ __align__(32) _Float16 vtileT[2][DHEAD * 32];// [d][key], double buffered

    const int tid  = threadIdx.x;
    const int w    = tid >> 5;
    const int lane = tid & 31;
    const int bb   = blockIdx.z;
    const int h    = blockIdx.y;
    const int r0   = blockIdx.x * 128;
    const int m0   = r0 + w * 16;           // wave's query base
    const int mr   = lane & 15;
    const int hi   = lane >> 4;
    const int kba  = hi ? 16 : 0;           // A-fragment K base
    const int kbb  = hi ? 16 : 0;           // B-fragment K base
    const int gq   = m0 + mr;               // this lane's query row

    const _Float16* kbase = kv + (size_t)(bb * SEQ) * KVLD;
    const _Float16* vbase = kbase + DHEAD;

    // Q^T B-fragments with fused l2norm (one shuffle completes ||q||^2)
    const _Float16* qrow = q + (size_t)(bb * SEQ + gq) * DIM + h * DHEAD;
    v16h qb[2];
#pragma unroll
    for (int t = 0; t < 2; ++t) qb[t] = *(const v16h*)(qrow + t * 32 + kbb);
    {
        float ssq = 0.f;
#pragma unroll
        for (int t = 0; t < 2; ++t)
#pragma unroll
            for (int i = 0; i < 16; ++i) {
                float v = (float)qb[t][i];
                ssq += v * v;
            }
        ssq += __shfl_xor(ssq, 16);
        const float qs = 4.0f / fmaxf(sqrtf(ssq), 1e-12f);
#pragma unroll
        for (int t = 0; t < 2; ++t)
#pragma unroll
            for (int i = 0; i < 16; ++i)
                qb[t][i] = (_Float16)((float)qb[t][i] * qs);
    }

    v8f acc[4] = {};                        // O^T tiles over d
    float mrun = -1e30f, lrun = 0.f;

    const int kend = r0 + 128;              // uniform across the block
    int buf = 0;
    stage_kv(kbase, vbase, 0, ktile[0], vtileT[0], tid);

    for (int j0 = 0; j0 < kend; j0 += 32) {
        asm volatile("s_wait_asynccnt 0x0" ::: "memory");
        __syncthreads();
        if (j0 + 32 < kend)                 // uniform branch
            stage_kv(kbase, vbase, j0 + 32, ktile[buf ^ 1], vtileT[buf ^ 1], tid);

        // ---- S^T: hoist all 4 K A-fragments, then 4 WMMAs back-to-back ----
        v16h ka[2][2];
#pragma unroll
        for (int mt = 0; mt < 2; ++mt)
#pragma unroll
            for (int kt = 0; kt < 2; ++kt) {
                h8v klo = *(const h8v*)&ktile[buf][(mt * 16 + mr) * DHEAD + kt * 32 + kba];
                h8v khi = *(const h8v*)&ktile[buf][(mt * 16 + mr) * DHEAD + kt * 32 + kba + 8];
#pragma unroll
                for (int i = 0; i < 8; ++i) { ka[mt][kt][i] = klo[i]; ka[mt][kt][i + 8] = khi[i]; }
            }
        v8f st[2] = {};
#pragma unroll
        for (int mt = 0; mt < 2; ++mt)
#pragma unroll
            for (int kt = 0; kt < 2; ++kt)
                st[mt] = __builtin_amdgcn_wmma_f32_16x16x32_f16(
                    false, ka[mt][kt], false, qb[kt], (short)0, st[mt], false, false);

        // causal mask (only diagonal-region tiles; wave-uniform skip otherwise)
        float xv[16];
        if (j0 + 31 > m0) {
#pragma unroll
            for (int mt = 0; mt < 2; ++mt)
#pragma unroll
                for (int j = 0; j < 8; ++j) {
                    const int key = j0 + mt * 16 + j + hi * 8;
                    xv[mt * 8 + j] = (key > gq) ? -1e30f : st[mt][j];
                }
        } else {
#pragma unroll
            for (int mt = 0; mt < 2; ++mt)
#pragma unroll
                for (int j = 0; j < 8; ++j) xv[mt * 8 + j] = st[mt][j];
        }

        // online softmax for this lane's query
        float mx = xv[0];
#pragma unroll
        for (int i = 1; i < 16; ++i) mx = fmaxf(mx, xv[i]);
        mx = fmaxf(mx, __shfl_xor(mx, 16));
        const float mnew = fmaxf(mrun, mx);
        float rs = 0.f;
        h8v p0, p1;
#pragma unroll
        for (int i = 0; i < 8; ++i) {
            float p = __expf(xv[i] - mnew);
            rs += p;
            p0[i] = (_Float16)p;
        }
#pragma unroll
        for (int i = 0; i < 8; ++i) {
            float p = __expf(xv[8 + i] - mnew);
            rs += p;
            p1[i] = (_Float16)p;
        }
        rs += __shfl_xor(rs, 16);
        const float sc = __expf(mrun - mnew);
        lrun = lrun * sc + rs;
        mrun = mnew;
#pragma unroll
        for (int t = 0; t < 4; ++t) acc[t] = acc[t] * sc;

        // P^T to per-wave LDS: [query][key]; lane's keys contiguous per tile
        *(h8v*)&pls[w][mr * 32 + 0  + 8 * hi] = p0;
        *(h8v*)&pls[w][mr * 32 + 16 + 8 * hi] = p1;

        // ---- P.V: hoist P B-fragment + all 4 V^T A-fragments, then 4 WMMAs ----
        v16h pb = *(const v16h*)&pls[w][mr * 32 + kbb];
        v16h va[4];
#pragma unroll
        for (int t = 0; t < 4; ++t) {
            h8v vlo = *(const h8v*)&vtileT[buf][(t * 16 + mr) * 32 + kba];
            h8v vhi = *(const h8v*)&vtileT[buf][(t * 16 + mr) * 32 + kba + 8];
#pragma unroll
            for (int i = 0; i < 8; ++i) { va[t][i] = vlo[i]; va[t][i + 8] = vhi[i]; }
        }
#pragma unroll
        for (int t = 0; t < 4; ++t)
            acc[t] = __builtin_amdgcn_wmma_f32_16x16x32_f16(
                false, va[t], false, pb, (short)0, acc[t], false, false);
        buf ^= 1;
    }

    // epilogue: O^T element (d = t*16 + j + 8*hi, query = mr); d contiguous in j
    const float inv = 1.0f / lrun;
    _Float16* orow = o + (size_t)(bb * SEQ + gq) * DIM + h * DHEAD;
#pragma unroll
    for (int t = 0; t < 4; ++t) {
        h8v ov;
#pragma unroll
        for (int j = 0; j < 8; ++j) ov[j] = (_Float16)(acc[t][j] * inv);
        *(h8v*)(orow + t * 16 + 8 * hi) = ov;
    }
}

// ---------------------------------------------------------------------------
extern "C" void kernel_launch(void* const* d_in, const int* in_sizes, int n_in,
                              void* d_out, int out_size, void* d_ws, size_t ws_size,
                              hipStream_t stream) {
    const float* x    = (const float*)d_in[0];
    const float* g1   = (const float*)d_in[1];
    const float* Wq   = (const float*)d_in[2];
    const float* Wkv  = (const float*)d_in[3];
    const float* Wout = (const float*)d_in[4];
    const float* g2   = (const float*)d_in[5];
    float* out = (float*)d_out;

    char* ws = (char*)d_ws;
    _Float16* xn_h   = (_Float16*)(ws);                       //  8 MB
    _Float16* wq_h   = (_Float16*)(ws + 8388608);             //  2 MB
    _Float16* wkv_h  = (_Float16*)(ws + 10485760);            //  256 KB
    _Float16* wout_h = (_Float16*)(ws + 10747904);            //  2 MB
    _Float16* q_h    = (_Float16*)(ws + 12845056);            //  8 MB
    _Float16* kv_h   = (_Float16*)(ws + 21233664);            //  1 MB
    _Float16* o_h    = (_Float16*)(ws + 22282240);            //  8 MB
    float*    proj_f = (float*)   (ws + 30670848);            // 16 MB

    ln_rows<_Float16><<<ROWS, 256, 0, stream>>>(x, g1, xn_h);

    f32_to_f16<<<(DIM * DIM + 255) / 256, 256, 0, stream>>>(Wq, wq_h, DIM * DIM);
    f32_to_f16<<<(KVLD * DIM + 255) / 256, 256, 0, stream>>>(Wkv, wkv_h, KVLD * DIM);
    f32_to_f16<<<(DIM * DIM + 255) / 256, 256, 0, stream>>>(Wout, wout_h, DIM * DIM);

    gemm_nt_wmma<_Float16><<<dim3(DIM / 64, ROWS / 128), 256, 0, stream>>>(
        xn_h, wq_h, q_h, ROWS, DIM, DIM);
    gemm_nt_wmma<_Float16><<<dim3(KVLD / 64, ROWS / 128), 256, 0, stream>>>(
        xn_h, wkv_h, kv_h, ROWS, KVLD, DIM);

    l2norm_k<<<ROWS / 8, 256, 0, stream>>>(kv_h);

    flash_attn<<<dim3(SEQ / 128, HEADS, BATCH), 256, 0, stream>>>(q_h, kv_h, o_h);

    gemm_nt_wmma<float><<<dim3(DIM / 64, ROWS / 128), 256, 0, stream>>>(
        o_h, wout_h, proj_f, ROWS, DIM, DIM);
    ln_rows<float><<<ROWS, 256, 0, stream>>>(proj_f, g2, out);
}
